// CasualConvTran_74869869904302
// MI455X (gfx1250) — matrix operand order repaired and checked by reading, not compiled, not run on hardware
//
#include <hip/hip_runtime.h>
#include <hip/hip_bf16.h>
#include <math.h>

typedef __bf16 bf16_t;
typedef __attribute__((ext_vector_type(16))) __bf16 v16bf;
typedef __attribute__((ext_vector_type(8)))  __bf16 v8bf;
typedef __attribute__((ext_vector_type(8)))  float  v8f;

static __device__ __forceinline__ int laneid() { return threadIdx.x & 31; }
static __device__ __forceinline__ int waveid() { return threadIdx.x >> 5; }

static __device__ __forceinline__ v8f wmma_bf16(v16bf a, v16bf b, v8f c) {
  return __builtin_amdgcn_wmma_f32_16x16x32_bf16(false, a, false, b, (short)0, c, false, false);
}

static __device__ __forceinline__ v16bf concat8(v8bf lo, v8bf hi) {
  return __builtin_shufflevector(lo, hi, 0, 1, 2, 3, 4, 5, 6, 7, 8, 9, 10, 11, 12, 13, 14, 15);
}
static __device__ __forceinline__ v8bf zero8() {
  v8bf z = {};
  return z;
}

// ---- Fragment loaders -------------------------------------------------------
// Pre-swizzled B fragment: frag[(fragIdx*32 + lane)*16 + i] ; two b128 loads.
static __device__ __forceinline__ v16bf loadBfrag(const bf16_t* __restrict__ frag, int fragIdx) {
  const v8bf* p = (const v8bf*)(frag + ((size_t)fragIdx * 32 + laneid()) * 16);
  return concat8(p[0], p[1]);
}

// A fragment from row-major LDS (stride elements, 16B-aligned runs): two ds_load_b128.
static __device__ __forceinline__ v16bf loadA_lds(const bf16_t* base, int mb, int stride, int kb0) {
  int lane = laneid();
  int m = mb + (lane & 15);
  int ko = (lane >> 4) << 4;            // 0 or 16
  const bf16_t* r = base + m * stride + kb0 + ko;
  return concat8(*(const v8bf*)(r), *(const v8bf*)(r + 8));
}

static __device__ __forceinline__ float gelu_exact(float z) {
  return 0.5f * z * (1.0f + erff(z * 0.70710678118654752f));
}

// ---------------------------------------------------------------------------
// Kernel 0: weight repack into fragment-swizzled bf16 B-layouts + BN fold + tAPE.
// Fragment (kc,nt): lane l holds col n=nt*16+(l&15), K = kc*32 + ((l>>4)<<4) + i.
// ---------------------------------------------------------------------------
__global__ void prep_kernel(
    const float* W1, const float* b1, const float* g1, const float* be1, const float* m1, const float* v1,
    const float* W2, const float* b2, const float* g2, const float* be2, const float* m2, const float* v2,
    const float* W3, const float* b3, const float* g3, const float* be3, const float* m3, const float* v3,
    const float* Wq, const float* Wk, const float* Wv,
    const float* Wf1, const float* Wf2,
    bf16_t* Bw1, bf16_t* Bw2, bf16_t* Bw3,
    bf16_t* Bwq, bf16_t* Bwk, bf16_t* Bwv,
    bf16_t* Bwf1, bf16_t* Bwf2,
    float* scale1, float* shift1, float* scale2, float* shift2, float* scale3, float* shift3,
    float* pe)
{
  int tid = blockIdx.x * blockDim.x + threadIdx.x;
  int nth = gridDim.x * blockDim.x;

  // Bw1: Ktot=32 (K=ci*8+k), NT=8.  W1 flat co*32 + K.
  for (int i = tid; i < 1 * 8 * 512; i += nth) {
    int elem = i & 15, lane = (i >> 4) & 31, ft = i >> 9;
    int nt = ft % 8;
    int n = nt * 16 + (lane & 15);
    int K = ((lane >> 4) << 4) + elem;
    Bw1[i] = (bf16_t)W1[n * 32 + K];
  }
  // Bw2: Ktot=640 (K=k*128+ci), NT=8.  W2 flat co*640 + ci*5 + k.
  for (int i = tid; i < 20 * 8 * 512; i += nth) {
    int elem = i & 15, lane = (i >> 4) & 31, ft = i >> 9;
    int kc = ft / 8, nt = ft % 8;
    int n = nt * 16 + (lane & 15);
    int K = kc * 32 + ((lane >> 4) << 4) + elem;
    int k = K >> 7, ci = K & 127;
    Bw2[i] = (bf16_t)W2[n * 640 + ci * 5 + k];
  }
  // Bw3: Ktot=384, NT=8.  W3 flat co*384 + ci*3 + k.
  for (int i = tid; i < 12 * 8 * 512; i += nth) {
    int elem = i & 15, lane = (i >> 4) & 31, ft = i >> 9;
    int kc = ft / 8, nt = ft % 8;
    int n = nt * 16 + (lane & 15);
    int K = kc * 32 + ((lane >> 4) << 4) + elem;
    int k = K >> 7, ci = K & 127;
    Bw3[i] = (bf16_t)W3[n * 384 + ci * 3 + k];
  }
  // Bwq/Bwk/Bwv: Ktot=128, NT=8.
  for (int i = tid; i < 4 * 8 * 512; i += nth) {
    int elem = i & 15, lane = (i >> 4) & 31, ft = i >> 9;
    int kc = ft / 8, nt = ft % 8;
    int n = nt * 16 + (lane & 15);
    int K = kc * 32 + ((lane >> 4) << 4) + elem;
    Bwq[i] = (bf16_t)Wq[n * 128 + K];
    Bwk[i] = (bf16_t)Wk[n * 128 + K];
    Bwv[i] = (bf16_t)Wv[n * 128 + K];
  }
  // Bwf1: Ktot=128, NT=32.
  for (int i = tid; i < 4 * 32 * 512; i += nth) {
    int elem = i & 15, lane = (i >> 4) & 31, ft = i >> 9;
    int kc = ft / 32, nt = ft % 32;
    int n = nt * 16 + (lane & 15);
    int K = kc * 32 + ((lane >> 4) << 4) + elem;
    Bwf1[i] = (bf16_t)Wf1[n * 128 + K];
  }
  // Bwf2: Ktot=512, NT=8.
  for (int i = tid; i < 16 * 8 * 512; i += nth) {
    int elem = i & 15, lane = (i >> 4) & 31, ft = i >> 9;
    int kc = ft / 8, nt = ft % 8;
    int n = nt * 16 + (lane & 15);
    int K = kc * 32 + ((lane >> 4) << 4) + elem;
    Bwf2[i] = (bf16_t)Wf2[n * 512 + K];
  }
  for (int c = tid; c < 128; c += nth) {
    float s;
    s = g1[c] * rsqrtf(v1[c] + 1e-5f); scale1[c] = s; shift1[c] = (b1[c] - m1[c]) * s + be1[c];
    s = g2[c] * rsqrtf(v2[c] + 1e-5f); scale2[c] = s; shift2[c] = (b2[c] - m2[c]) * s + be2[c];
    s = g3[c] * rsqrtf(v3[c] + 1e-5f); scale3[c] = s; shift3[c] = (b3[c] - m3[c]) * s + be3[c];
  }
  for (int i = tid; i < 32 * 128; i += nth) {   // tAPE
    int t = i >> 7, e = i & 127;
    int pair = e >> 1;
    float div = __expf((float)(2 * pair) * (-logf(10000.0f) / 128.0f));
    float ang = (float)t * div * 4.0f;          // * (EMB/T)
    pe[i] = (e & 1) ? __cosf(ang) : __sinf(ang);
  }
}

// ---------------------------------------------------------------------------
// Kernel 1: conv1 (4ch,k=8,s=2,d=1) + BN + GELU.  out h1: [b][l:128][co:128] bf16
// ---------------------------------------------------------------------------
__global__ void __launch_bounds__(256) conv1_kernel(
    const float* __restrict__ x, const bf16_t* __restrict__ Bw1,
    const float* __restrict__ scale1, const float* __restrict__ shift1,
    bf16_t* __restrict__ h1)
{
  __shared__ __align__(16) float xl[4 * 256];
  int b = blockIdx.x;
  {
    const uint4* s = (const uint4*)(x + (size_t)b * 1024);
    uint4* d = (uint4*)xl;
    for (int i = threadIdx.x; i < 256; i += 256) d[i] = s[i];
  }
  __syncthreads();
  int w = waveid(), lane = laneid();
  int mb = w * 16;                 // 8 m-tiles cover 128 output positions
  // elementwise A build (float source, one-time per wave)
  v16bf a;
  {
    int m = lane & 15;
    int kb = (lane >> 4) << 4;
#pragma unroll
    for (int half = 0; half < 2; ++half) {
#pragma unroll
      for (int i = 0; i < 8; ++i) {
        int k = kb + half * 8 + i;
        int ci = k >> 3, kk = k & 7;
        int pos = 2 * (mb + m) - 7 + kk;       // causal left pad 7
        a[half * 8 + i] = (bf16_t)((pos >= 0) ? xl[ci * 256 + pos] : 0.0f);
      }
    }
  }
  int col = lane & 15, row0 = (lane >> 4) * 8;
#pragma unroll
  for (int nt = 0; nt < 8; ++nt) {
    v16bf bb = loadBfrag(Bw1, nt);
    v8f c = {};
    c = wmma_bf16(a, bb, c);
    int co = nt * 16 + col;
    float sc = scale1[co], sh = shift1[co];
#pragma unroll
    for (int r = 0; r < 8; ++r) {
      int l = mb + row0 + r;
      float g = gelu_exact(c[r] * sc + sh);
      h1[((size_t)b * 128 + l) * 128 + co] = (bf16_t)g;
    }
  }
}

// ---------------------------------------------------------------------------
// Kernel 2: conv2 (128ch,k=5,s=2,d=2) + BN + GELU.  out h2: [b][l:64][co:128]
// ---------------------------------------------------------------------------
__global__ void __launch_bounds__(256) conv2_kernel(
    const bf16_t* __restrict__ h1, const bf16_t* __restrict__ Bw2,
    const float* __restrict__ scale2, const float* __restrict__ shift2,
    bf16_t* __restrict__ h2)
{
  __shared__ __align__(16) bf16_t hl[128 * 128];
  int b = blockIdx.x;
  if (threadIdx.x == 0) __builtin_prefetch(Bw2, 0, 1);
  {
    const uint4* s = (const uint4*)(h1 + (size_t)b * 128 * 128);
    uint4* d = (uint4*)hl;
    for (int i = threadIdx.x; i < 128 * 128 / 8; i += 256) d[i] = s[i];
  }
  __syncthreads();
  int w = waveid(), lane = laneid();
  int mb = (w & 3) * 16;           // 4 m-tiles (64 rows)
  int nbase = (w >> 2) * 4;        // 4 n-tiles per wave
  int m = lane & 15, ko = (lane >> 4) << 4;
  v8f acc[4] = {};
#pragma unroll
  for (int kc = 0; kc < 20; ++kc) {      // K = 5*128 = 640
    int kb0 = kc * 32;
    int tap = kb0 >> 7;                  // conv tap (constant within chunk)
    int cibase = kb0 & 127;
    int pos = 2 * (mb + m) - 8 + 2 * tap;   // causal pad 8, dilation 2
    v16bf a;
    if (pos >= 0) {
      const bf16_t* r = hl + pos * 128 + cibase + ko;
      a = concat8(*(const v8bf*)(r), *(const v8bf*)(r + 8));
    } else {
      a = concat8(zero8(), zero8());
    }
#pragma unroll
    for (int j = 0; j < 4; ++j) {
      v16bf bb = loadBfrag(Bw2, kc * 8 + nbase + j);
      acc[j] = wmma_bf16(a, bb, acc[j]);
    }
  }
  int col = lane & 15, row0 = (lane >> 4) * 8;
#pragma unroll
  for (int j = 0; j < 4; ++j) {
    int co = (nbase + j) * 16 + col;
    float sc = scale2[co], sh = shift2[co];
#pragma unroll
    for (int r = 0; r < 8; ++r) {
      int l = mb + row0 + r;
      float g = gelu_exact(acc[j][r] * sc + sh);
      h2[((size_t)b * 64 + l) * 128 + co] = (bf16_t)g;
    }
  }
}

// ---------------------------------------------------------------------------
// Kernel 3: conv3 (128ch,k=3,s=2,d=2) + BN + GELU + tAPE.  out xpos: [b][32][128]
// ---------------------------------------------------------------------------
__global__ void __launch_bounds__(256) conv3_kernel(
    const bf16_t* __restrict__ h2, const bf16_t* __restrict__ Bw3,
    const float* __restrict__ scale3, const float* __restrict__ shift3,
    const float* __restrict__ pe, bf16_t* __restrict__ xpos)
{
  __shared__ __align__(16) bf16_t hl[64 * 128];
  int b = blockIdx.x;
  {
    const uint4* s = (const uint4*)(h2 + (size_t)b * 64 * 128);
    uint4* d = (uint4*)hl;
    for (int i = threadIdx.x; i < 64 * 128 / 8; i += 256) d[i] = s[i];
  }
  __syncthreads();
  int w = waveid(), lane = laneid();
  int mb = (w & 1) * 16;           // 2 m-tiles (32 rows)
  int nts0 = (w >> 1);             // 2 n-tiles per wave: nts0, nts0+4
  int m = lane & 15, ko = (lane >> 4) << 4;
  v8f acc[2] = {};
#pragma unroll
  for (int kc = 0; kc < 12; ++kc) {      // K = 3*128 = 384
    int kb0 = kc * 32;
    int tap = kb0 >> 7;
    int cibase = kb0 & 127;
    int pos = 2 * (mb + m) - 4 + 2 * tap;   // causal pad 4, dilation 2
    v16bf a;
    if (pos >= 0) {
      const bf16_t* r = hl + pos * 128 + cibase + ko;
      a = concat8(*(const v8bf*)(r), *(const v8bf*)(r + 8));
    } else {
      a = concat8(zero8(), zero8());
    }
#pragma unroll
    for (int j = 0; j < 2; ++j) {
      v16bf bb = loadBfrag(Bw3, kc * 8 + nts0 + 4 * j);
      acc[j] = wmma_bf16(a, bb, acc[j]);
    }
  }
  int col = lane & 15, row0 = (lane >> 4) * 8;
#pragma unroll
  for (int j = 0; j < 2; ++j) {
    int co = (nts0 + 4 * j) * 16 + col;
    float sc = scale3[co], sh = shift3[co];
#pragma unroll
    for (int r = 0; r < 8; ++r) {
      int l = mb + row0 + r;
      float g = gelu_exact(acc[j][r] * sc + sh);
      xpos[((size_t)b * 32 + l) * 128 + co] = (bf16_t)(g + pe[l * 128 + co]);
    }
  }
}

// ---------------------------------------------------------------------------
// Kernel 4: attention + LNs + FFN + mean over T.  One workgroup per batch.
// 64KB LDS, manually carved and phase-aliased.
// ---------------------------------------------------------------------------
__global__ void __launch_bounds__(256) attn_kernel(
    const bf16_t* __restrict__ xpos,
    const bf16_t* __restrict__ Bwq, const bf16_t* __restrict__ Bwk, const bf16_t* __restrict__ Bwv,
    const float* __restrict__ rel_bias,
    const float* __restrict__ g_to, const float* __restrict__ b_to,
    const float* __restrict__ g_ln1, const float* __restrict__ b_ln1,
    const float* __restrict__ g_ln2, const float* __restrict__ b_ln2,
    const bf16_t* __restrict__ Bwf1, const float* __restrict__ bf1v,
    const bf16_t* __restrict__ Bwf2, const float* __restrict__ bf2v,
    const float* __restrict__ pe, float* __restrict__ out)
{
  __shared__ __align__(16) char smem[65536];
  bf16_t* xp   = (bf16_t*)(smem);             // [0,8K)    xpos tile (phases 1-4)
  bf16_t* qs   = (bf16_t*)(smem + 8192);      // [8K,16K)  Q         (phases 2-3)
  bf16_t* ks   = (bf16_t*)(smem + 16384);     // [16K,24K) K         (phases 2-3)
  bf16_t* vt   = (bf16_t*)(smem + 24576);     // [24K,32K) V^T [d:128][t:32] (2-3)
  bf16_t* scws = (bf16_t*)(smem + 32768);     // [32K,40K) per-wave attn probs (3)
  float*  os   = (float*)(smem + 40960);      // [40K,56K) attn out f32 (3-4)
  bf16_t* attb = (bf16_t*)(smem + 8192);      // alias qs  (phases 4-6)
  bf16_t* ffb  = (bf16_t*)(smem + 16384);     // [16K,48K) FFN hidden (5-6)
  float*  yb   = (float*)(smem + 49152);      // [48K,64K) FFN2+res / LN2 (6)

  int b = blockIdx.x;
  int w = waveid(), lane = laneid();
  int col = lane & 15, row0 = (lane >> 4) * 8;

  {
    const uint4* s = (const uint4*)(xpos + (size_t)b * 32 * 128);
    uint4* d = (uint4*)xp;
    for (int i = threadIdx.x; i < 32 * 128 / 8; i += 256) d[i] = s[i];
  }
  __syncthreads();

  // ---- Phase 2: Q,K,V = xp @ W^T (48 tiles, K=128). q pre-scaled, V also transposed.
  const float qscale = 0.0883883476483184f;   // EMB^-0.5
  for (int t = w; t < 48; t += 8) {
    int mat = t / 16, rem = t % 16;
    int mb = (rem >> 3) * 16, nt = rem & 7, nb = nt * 16;
    const bf16_t* Bm = (mat == 0) ? Bwq : (mat == 1) ? Bwk : Bwv;
    v8f c = {};
#pragma unroll
    for (int kc = 0; kc < 4; ++kc) {
      v16bf a  = loadA_lds(xp, mb, 128, kc * 32);
      v16bf bb = loadBfrag(Bm, kc * 8 + nt);
      c = wmma_bf16(a, bb, c);
    }
    if (mat == 0) {
#pragma unroll
      for (int r = 0; r < 8; ++r) qs[(mb + row0 + r) * 128 + nb + col] = (bf16_t)(c[r] * qscale);
    } else if (mat == 1) {
#pragma unroll
      for (int r = 0; r < 8; ++r) ks[(mb + row0 + r) * 128 + nb + col] = (bf16_t)c[r];
    } else {
#pragma unroll
      for (int r = 0; r < 8; ++r) vt[(nb + col) * 32 + mb + row0 + r] = (bf16_t)c[r];
    }
  }
  __syncthreads();

  // ---- Phase 3: per-head scores, softmax, +bias (after softmax!), @V ----
  bf16_t* mysc = scws + w * 512;              // 16x32 probs
  for (int cb = w; cb < 16; cb += 8) {
    int h = cb & 7, mb = (cb >> 3) * 16, hb = h * 16;
    // A = q rows, K padded 16->32: lanes<16 contiguous K 0..15, lanes>=16 zero
    v16bf aq;
    {
      int m = mb + (lane & 15);
      const bf16_t* r = qs + m * 128 + hb;
      if (lane < 16) aq = concat8(*(const v8bf*)(r), *(const v8bf*)(r + 8));
      else aq = concat8(zero8(), zero8());
    }
    // B tiles = K^T, K padded: lanes>=16 (K 16..31) are zero
    v8f c0 = {}, c1 = {};
    {
      v16bf b0, b1;
      if (lane < 16) {
        const bf16_t* r0 = ks + (lane & 15) * 128 + hb;
        const bf16_t* r1 = ks + (16 + (lane & 15)) * 128 + hb;
        b0 = concat8(*(const v8bf*)(r0), *(const v8bf*)(r0 + 8));
        b1 = concat8(*(const v8bf*)(r1), *(const v8bf*)(r1 + 8));
      } else {
        b0 = concat8(zero8(), zero8());
        b1 = b0;
      }
      c0 = wmma_bf16(aq, b0, c0);
      c1 = wmma_bf16(aq, b1, c1);
    }
#pragma unroll
    for (int r = 0; r < 8; ++r) {            // row softmax across 16 lanes x 2 regs
      float v0 = c0[r], v1 = c1[r];
      float mx = fmaxf(v0, v1);
      for (int msk = 1; msk <= 8; msk <<= 1) mx = fmaxf(mx, __shfl_xor(mx, msk, 32));
      float e0 = __expf(v0 - mx), e1 = __expf(v1 - mx);
      float s = e0 + e1;
      for (int msk = 1; msk <= 8; msk <<= 1) s += __shfl_xor(s, msk, 32);
      float inv = 1.0f / s;
      int ig = mb + row0 + r;
      float p0 = e0 * inv + rel_bias[(ig - col + 31) * 8 + h];
      float p1 = e1 * inv + rel_bias[(ig - (16 + col) + 31) * 8 + h];
      mysc[(row0 + r) * 32 + col] = (bf16_t)p0;
      mysc[(row0 + r) * 32 + 16 + col] = (bf16_t)p1;
    }
    // attn @ V  (A: probs 16x32 contiguous; B: from V^T contiguous)
    v16bf ap = loadA_lds(mysc, 0, 32, 0);
    v16bf bv;
    {
      int n = lane & 15, kb = (lane >> 4) << 4;
      const bf16_t* r = vt + (hb + n) * 32 + kb;
      bv = concat8(*(const v8bf*)(r), *(const v8bf*)(r + 8));
    }
    v8f co_ = {};
    co_ = wmma_bf16(ap, bv, co_);
#pragma unroll
    for (int r = 0; r < 8; ++r) os[(mb + row0 + r) * 128 + hb + col] = co_[r];
  }
  __syncthreads();

  // ---- Phase 4: LN(to_out), residual with x_src = xp - pe, LN1 -> attb ----
  for (int rr = 0; rr < 4; ++rr) {
    int t = w * 4 + rr;
    float vals[4], mu = 0.0f;
#pragma unroll
    for (int j = 0; j < 4; ++j) { vals[j] = os[t * 128 + lane + 32 * j]; mu += vals[j]; }
    for (int msk = 1; msk <= 16; msk <<= 1) mu += __shfl_xor(mu, msk, 32);
    mu *= (1.0f / 128.0f);
    float var = 0.0f;
#pragma unroll
    for (int j = 0; j < 4; ++j) { float d = vals[j] - mu; var += d * d; }
    for (int msk = 1; msk <= 16; msk <<= 1) var += __shfl_xor(var, msk, 32);
    float inv = rsqrtf(var * (1.0f / 128.0f) + 1e-5f);
    float s1[4], mu2 = 0.0f;
#pragma unroll
    for (int j = 0; j < 4; ++j) {
      int c = lane + 32 * j;
      float oln = (vals[j] - mu) * inv * g_to[c] + b_to[c];
      float xsrc = (float)xp[t * 128 + c] - pe[t * 128 + c];
      s1[j] = xsrc + oln; mu2 += s1[j];
    }
    for (int msk = 1; msk <= 16; msk <<= 1) mu2 += __shfl_xor(mu2, msk, 32);
    mu2 *= (1.0f / 128.0f);
    float var2 = 0.0f;
#pragma unroll
    for (int j = 0; j < 4; ++j) { float d = s1[j] - mu2; var2 += d * d; }
    for (int msk = 1; msk <= 16; msk <<= 1) var2 += __shfl_xor(var2, msk, 32);
    float inv2 = rsqrtf(var2 * (1.0f / 128.0f) + 1e-5f);
#pragma unroll
    for (int j = 0; j < 4; ++j) {
      int c = lane + 32 * j;
      attb[t * 128 + c] = (bf16_t)((s1[j] - mu2) * inv2 * g_ln1[c] + b_ln1[c]);
    }
  }
  __syncthreads();

  // ---- Phase 5: FFN1 = relu(att @ Wf1^T + bf1) -> ffb (32x512 bf16) ----
  for (int t = w; t < 64; t += 8) {
    int mb = (t & 1) * 16, nt = t >> 1, nb = nt * 16;
    v8f c = {};
#pragma unroll
    for (int kc = 0; kc < 4; ++kc) {
      v16bf a  = loadA_lds(attb, mb, 128, kc * 32);
      v16bf bb = loadBfrag(Bwf1, kc * 32 + nt);
      c = wmma_bf16(a, bb, c);
    }
#pragma unroll
    for (int r = 0; r < 8; ++r) {
      int cc = nb + col;
      ffb[(mb + row0 + r) * 512 + cc] = (bf16_t)fmaxf(c[r] + bf1v[cc], 0.0f);
    }
  }
  __syncthreads();

  // ---- Phase 6a: FFN2 + bias + residual(att) -> yb f32 ----
  for (int t = w; t < 16; t += 8) {
    int mb = (t & 1) * 16, nt = t >> 1, nb = nt * 16;
    v8f c = {};
#pragma unroll
    for (int kc = 0; kc < 16; ++kc) {
      v16bf a  = loadA_lds(ffb, mb, 512, kc * 32);
      v16bf bb = loadBfrag(Bwf2, kc * 8 + nt);
      c = wmma_bf16(a, bb, c);
    }
#pragma unroll
    for (int r = 0; r < 8; ++r) {
      int cc = nb + col, rw = mb + row0 + r;
      yb[rw * 128 + cc] = c[r] + bf2v[cc] + (float)attb[rw * 128 + cc];
    }
  }
  __syncthreads();

  // ---- Phase 6b: LN2 per row (in place) ----
  for (int rr = 0; rr < 4; ++rr) {
    int t = w * 4 + rr;
    float vals[4], mu = 0.0f;
#pragma unroll
    for (int j = 0; j < 4; ++j) { vals[j] = yb[t * 128 + lane + 32 * j]; mu += vals[j]; }
    for (int msk = 1; msk <= 16; msk <<= 1) mu += __shfl_xor(mu, msk, 32);
    mu *= (1.0f / 128.0f);
    float var = 0.0f;
#pragma unroll
    for (int j = 0; j < 4; ++j) { float d = vals[j] - mu; var += d * d; }
    for (int msk = 1; msk <= 16; msk <<= 1) var += __shfl_xor(var, msk, 32);
    float inv = rsqrtf(var * (1.0f / 128.0f) + 1e-5f);
#pragma unroll
    for (int j = 0; j < 4; ++j) {
      int c = lane + 32 * j;
      yb[t * 128 + c] = (vals[j] - mu) * inv * g_ln2[c] + b_ln2[c];
    }
  }
  __syncthreads();

  // ---- Phase 7: mean over T ----
  for (int c = threadIdx.x; c < 128; c += 256) {
    float s = 0.0f;
    for (int t = 0; t < 32; ++t) s += yb[t * 128 + c];
    out[(size_t)b * 128 + c] = s * (1.0f / 32.0f);
  }
}

// ---------------------------------------------------------------------------
extern "C" void kernel_launch(void* const* d_in, const int* in_sizes, int n_in,
                              void* d_out, int out_size, void* d_ws, size_t ws_size,
                              hipStream_t stream) {
  (void)in_sizes; (void)n_in; (void)out_size; (void)ws_size;
  const float* x   = (const float*)d_in[0];
  const float* W1  = (const float*)d_in[1];
  const float* b1  = (const float*)d_in[2];
  const float* g1  = (const float*)d_in[3];
  const float* be1 = (const float*)d_in[4];
  const float* m1  = (const float*)d_in[5];
  const float* v1  = (const float*)d_in[6];
  const float* W2  = (const float*)d_in[7];
  const float* b2  = (const float*)d_in[8];
  const float* g2  = (const float*)d_in[9];
  const float* be2 = (const float*)d_in[10];
  const float* m2  = (const float*)d_in[11];
  const float* v2  = (const float*)d_in[12];
  const float* W3  = (const float*)d_in[13];
  const float* b3  = (const float*)d_in[14];
  const float* g3  = (const float*)d_in[15];
  const float* be3 = (const float*)d_in[16];
  const float* m3  = (const float*)d_in[17];
  const float* v3  = (const float*)d_in[18];
  const float* Wq  = (const float*)d_in[19];
  const float* Wk  = (const float*)d_in[20];
  const float* Wv  = (const float*)d_in[21];
  const float* rel = (const float*)d_in[22];
  const float* gto = (const float*)d_in[23];
  const float* bto = (const float*)d_in[24];
  const float* gl1 = (const float*)d_in[25];
  const float* bl1 = (const float*)d_in[26];
  const float* gl2 = (const float*)d_in[27];
  const float* bl2 = (const float*)d_in[28];
  const float* Wf1 = (const float*)d_in[29];
  const float* bf1 = (const float*)d_in[30];
  const float* Wf2 = (const float*)d_in[31];
  const float* bf2 = (const float*)d_in[32];
  float* out = (float*)d_out;

  char* ws = (char*)d_ws;
  size_t off = 0;
  auto carve = [&](size_t bytes) -> char* {
    char* p = ws + off;
    off += (bytes + 255) & ~(size_t)255;
    return p;
  };
  bf16_t* Bw1  = (bf16_t*)carve((size_t)1 * 8 * 512 * 2);
  bf16_t* Bw2  = (bf16_t*)carve((size_t)20 * 8 * 512 * 2);
  bf16_t* Bw3  = (bf16_t*)carve((size_t)12 * 8 * 512 * 2);
  bf16_t* Bwq  = (bf16_t*)carve((size_t)4 * 8 * 512 * 2);
  bf16_t* Bwk  = (bf16_t*)carve((size_t)4 * 8 * 512 * 2);
  bf16_t* Bwv  = (bf16_t*)carve((size_t)4 * 8 * 512 * 2);
  bf16_t* Bwf1 = (bf16_t*)carve((size_t)4 * 32 * 512 * 2);
  bf16_t* Bwf2 = (bf16_t*)carve((size_t)16 * 8 * 512 * 2);
  float* scale1 = (float*)carve(128 * 4);
  float* shift1 = (float*)carve(128 * 4);
  float* scale2 = (float*)carve(128 * 4);
  float* shift2 = (float*)carve(128 * 4);
  float* scale3 = (float*)carve(128 * 4);
  float* shift3 = (float*)carve(128 * 4);
  float* pe   = (float*)carve(32 * 128 * 4);
  bf16_t* h1  = (bf16_t*)carve((size_t)2048 * 128 * 128 * 2);
  bf16_t* h2  = (bf16_t*)carve((size_t)2048 * 64 * 128 * 2);
  bf16_t* xps = (bf16_t*)carve((size_t)2048 * 32 * 128 * 2);

  prep_kernel<<<128, 256, 0, stream>>>(
      W1, b1, g1, be1, m1, v1, W2, b2, g2, be2, m2, v2, W3, b3, g3, be3, m3, v3,
      Wq, Wk, Wv, Wf1, Wf2, Bw1, Bw2, Bw3, Bwq, Bwk, Bwv, Bwf1, Bwf2,
      scale1, shift1, scale2, shift2, scale3, shift3, pe);
  conv1_kernel<<<2048, 256, 0, stream>>>(x, Bw1, scale1, shift1, h1);
  conv2_kernel<<<2048, 256, 0, stream>>>(h1, Bw2, scale2, shift2, h2);
  conv3_kernel<<<2048, 256, 0, stream>>>(h2, Bw3, scale3, shift3, pe, xps);
  attn_kernel<<<2048, 256, 0, stream>>>(
      xps, Bwq, Bwk, Bwv, rel, gto, bto, gl1, bl1, gl2, bl2,
      Bwf1, bf1, Bwf2, bf2, pe, out);
}
